// EquivariantProductBasisBlock_51900384805552
// MI455X (gfx1250) — matrix operand order repaired and checked by reading, not compiled, not run on hardware
//
#include <hip/hip_runtime.h>
#include <hip/hip_bf16.h>

// CDNA5 (gfx1250) fused equivariant-product-basis + dual GEMM.
// Phase A: fp32 pointwise basis expansion, staged to LDS as f16.
// Phase B: V_WMMA_F32_16X16X32_F16 (f16 inputs, fp32 accumulate) -> HBM-bound.

typedef _Float16 h8   __attribute__((ext_vector_type(8)));
typedef _Float16 v16h __attribute__((ext_vector_type(16)));
typedef float    v8f  __attribute__((ext_vector_type(8)));

#define FDIM 256
#define TM 32              // nodes (M rows) per workgroup
#define LDS_STRIDE 264     // row stride in f16 (528B: 16B-aligned, good bank spread)
#define INV_SQRT3 0.57735026918962576451f
#define OUT_SCALE 0.0625f  // 1/sqrt(256), folded into staged activations

// ---------------------------------------------------------------------------
// Pre-kernel: convert W0/W1 (fp32, k-major) -> f16, g-major (Wt[g*256+k]).
// Grid (16,16), block (16,16): LDS tile transpose, coalesced both ways.
// ---------------------------------------------------------------------------
__global__ void convert_W_kernel(const float* __restrict__ W0,
                                 const float* __restrict__ W1,
                                 _Float16* __restrict__ Wt0,
                                 _Float16* __restrict__ Wt1)
{
    __shared__ float t0[16][17];
    __shared__ float t1[16][17];
    const int tx = threadIdx.x, ty = threadIdx.y;
    const int bk = blockIdx.y * 16;   // k-tile
    const int bg = blockIdx.x * 16;   // g-tile
    t0[ty][tx] = W0[(bk + ty) * FDIM + (bg + tx)];
    t1[ty][tx] = W1[(bk + ty) * FDIM + (bg + tx)];
    __syncthreads();
    Wt0[(bg + ty) * FDIM + (bk + tx)] = (_Float16)t0[tx][ty];
    Wt1[(bg + ty) * FDIM + (bk + tx)] = (_Float16)t1[tx][ty];
}

// ---------------------------------------------------------------------------
// Main fused kernel.
// ---------------------------------------------------------------------------
__launch_bounds__(256)
__global__ void equiv_basis_gemm_kernel(
    const float* __restrict__ node_feats,   // (n, 256, 4) : s, vx, vy, vz
    const int*   __restrict__ node_specie,  // (n,)
    const float* __restrict__ w0,           // (10, 5, 256)
    const float* __restrict__ w1,           // (10, 4, 256)
    const _Float16* __restrict__ Wt0,       // (256, 256) f16, g-major
    const _Float16* __restrict__ Wt1,       // (256, 256) f16, g-major
    float* __restrict__ out,                // (n, 256, 4)
    int n)
{
    extern __shared__ _Float16 smem[];      // [4][TM][LDS_STRIDE] staged f16 activations
    const int tid  = threadIdx.x;
    const int base = blockIdx.x * TM;

    // ---------------- Phase A: pointwise basis expansion (fp32) -> LDS (f16) ----
    {
        const int f = tid;                  // 256 threads = 256 feature columns
        #pragma unroll 4
        for (int m = 0; m < TM; ++m) {
            const int node = base + m;
            float o0 = 0.f, o1 = 0.f, o2 = 0.f, o3 = 0.f;
            if (node < n) {
                const float4 nf = ((const float4*)node_feats)[node * FDIM + f];
                const int sp = node_specie[node];           // uniform across wave
                const float* c0 = w0 + (sp * 5) * FDIM + f; // tiny tables, cache resident
                const float* c1 = w1 + (sp * 4) * FDIM + f;
                const float s  = nf.x;
                const float vv = (nf.y * nf.y + nf.z * nf.z + nf.w * nf.w) * INV_SQRT3;
                const float s2 = s * s;
                const float out0 = c0[0] * s + c0[FDIM] * s2 + c0[2 * FDIM] * vv
                                 + c0[3 * FDIM] * (s2 * s) + c0[4 * FDIM] * (s * vv);
                const float cc   = c1[0] + c1[FDIM] * s + c1[2 * FDIM] * s2 + c1[3 * FDIM] * vv;
                o0 = out0 * OUT_SCALE;
                const float ccs = cc * OUT_SCALE;
                o1 = ccs * nf.y;
                o2 = ccs * nf.z;
                o3 = ccs * nf.w;
            }
            smem[(0 * TM + m) * LDS_STRIDE + f] = (_Float16)o0;
            smem[(1 * TM + m) * LDS_STRIDE + f] = (_Float16)o1;
            smem[(2 * TM + m) * LDS_STRIDE + f] = (_Float16)o2;
            smem[(3 * TM + m) * LDS_STRIDE + f] = (_Float16)o3;
        }
    }
    __syncthreads();

    // ---------------- Phase B: f16 WMMA GEMM, fp32 accumulate ------------------
    // V_WMMA_F32_16X16X32_F16 fragment layouts (wave32, half = lane>>4):
    //   A 16x32 : lane -> row M = lane&15; v16h = { K[8*half .. +8), K[16+8*half .. +8) }
    //   B 32x16 : lane -> col N = lane&15; v16h = { K[8*half .. +8), K[16+8*half .. +8) }  (g-major Wt!)
    //   C 16x16 : reg r -> M = r + 8*half, N = lane&15
    const int wave = tid >> 5;
    const int lane = tid & 31;
    const int half = lane >> 4;
    const int mk   = lane & 15;

    for (int jj = 0; jj < 2; ++jj) {
        const int j    = wave + jj * 8;     // N-tile (16 tiles of 16 cols)
        const int gcol = j * 16 + mk;

        v8f acc[2][4];
        #pragma unroll
        for (int mt = 0; mt < 2; ++mt)
            #pragma unroll
            for (int c = 0; c < 4; ++c)
                acc[mt][c] = {};

        const _Float16* __restrict__ b0p = Wt0 + gcol * FDIM + 8 * half;
        const _Float16* __restrict__ b1p = Wt1 + gcol * FDIM + 8 * half;

        #pragma unroll
        for (int k0 = 0; k0 < FDIM; k0 += 32) {
            // B fragments: two b128 loads each, K[8*half..+8) and K[16+8*half..+8) (L2-resident)
            union { v16h v; h8 h[2]; } b0, b1;
            b0.h[0] = *(const h8*)(b0p + k0);
            b0.h[1] = *(const h8*)(b0p + k0 + 16);
            b1.h[0] = *(const h8*)(b1p + k0);
            b1.h[1] = *(const h8*)(b1p + k0 + 16);

            #pragma unroll
            for (int mt = 0; mt < 2; ++mt) {
                #pragma unroll
                for (int c = 0; c < 4; ++c) {
                    const _Float16* ap =
                        smem + ((c * TM) + (mt * 16 + mk)) * LDS_STRIDE + k0;
                    union { v16h v; h8 h[2]; } a;
                    a.h[0] = *(const h8*)(ap + 8 * half);        // K chunk 0
                    a.h[1] = *(const h8*)(ap + 16 + 8 * half);   // K chunk 1
                    const v16h b = (c == 0) ? b0.v : b1.v;
                    acc[mt][c] = __builtin_amdgcn_wmma_f32_16x16x32_f16(
                        /*neg_a=*/false, a.v, /*neg_b=*/false, b,
                        /*c_mod=*/(short)0, acc[mt][c],
                        /*reuse_a=*/false, /*reuse_b=*/false);
                }
            }
        }

        // Coalesced float4 stores: all 4 output channels of (node, gcol) per lane.
        #pragma unroll
        for (int mt = 0; mt < 2; ++mt) {
            #pragma unroll
            for (int r = 0; r < 8; ++r) {
                const int node = base + mt * 16 + half * 8 + r;
                if (node < n) {
                    float4 o;
                    o.x = acc[mt][0][r];
                    o.y = acc[mt][1][r];
                    o.z = acc[mt][2][r];
                    o.w = acc[mt][3][r];
                    ((float4*)out)[node * FDIM + gcol] = o;
                }
            }
        }
    }
}

extern "C" void kernel_launch(void* const* d_in, const int* in_sizes, int n_in,
                              void* d_out, int out_size, void* d_ws, size_t ws_size,
                              hipStream_t stream) {
    const float* node_feats  = (const float*)d_in[0];
    const int*   node_specie = (const int*)d_in[1];
    const float* w0 = (const float*)d_in[2];
    const float* w1 = (const float*)d_in[3];
    const float* W0 = (const float*)d_in[4];
    const float* W1 = (const float*)d_in[5];
    float* out = (float*)d_out;

    // Workspace: f16 g-major copies of W0/W1 (128 KB each).
    _Float16* Wt0 = (_Float16*)d_ws;
    _Float16* Wt1 = Wt0 + FDIM * FDIM;

    convert_W_kernel<<<dim3(16, 16), dim3(16, 16), 0, stream>>>(W0, W1, Wt0, Wt1);

    const int n = in_sizes[1];                 // node count
    const int grid = (n + TM - 1) / TM;
    const size_t lds_bytes = 4u * TM * LDS_STRIDE * sizeof(_Float16); // 67,584 B

    equiv_basis_gemm_kernel<<<grid, 256, lds_bytes, stream>>>(
        node_feats, node_specie, w0, w1, Wt0, Wt1, out, n);
}
